// RqVae_29592324669839
// MI455X (gfx1250) — matrix operand order repaired and checked by reading, not compiled, not run on hardware
//
#include <hip/hip_runtime.h>
#include <stdint.h>

// ---------------------------------------------------------------------------
// RQ-VAE nearest-codebook quantization for MI455X (gfx1250, wave32, WMMA).
//   N=262144 points, K=1024 codes, D=64.
//   argmin_k ||x-c_k||^2  ==  argmax_k (x.c_k - 0.5*||c_k||^2)
// Compute-bound (AI ~170 FLOP/byte): v_wmma_f32_16x16x32_bf16 with hi/lo bf16
// split (hi*hi + hi*lo + lo*hi) for ~fp32 accuracy at bf16-matrix rate.
// Each wave: 2 row-tiles (32 rows) -> two interleaved WMMA chains (ILP) and
// 2x B-fragment reuse. 0.5*||c||^2 staged to LDS via async-load (ASYNCcnt).
// ---------------------------------------------------------------------------

typedef __attribute__((ext_vector_type(16))) __bf16 v16bf;
typedef __attribute__((ext_vector_type(8)))  float  v8f;
typedef __attribute__((ext_vector_type(4)))  float  v4f;
typedef __attribute__((ext_vector_type(2)))  float  v2f;

#define N_PTS   262144
#define K_CODES 1024
#define DIM     64

// ---------------------------------------------------------------------------
// Prep: codebook f32 [1024][64] -> bf16 hi plane, bf16 lo plane, 0.5*||c||^2.
// ---------------------------------------------------------------------------
__global__ void prep_codebook(const float* __restrict__ cb,
                              __bf16* __restrict__ hi,
                              __bf16* __restrict__ lo,
                              float* __restrict__ halfcsq) {
  int k = blockIdx.x * blockDim.x + threadIdx.x;
  if (k >= K_CODES) return;
  const float* row = cb + (size_t)k * DIM;
  float s = 0.f;
#pragma unroll 8
  for (int d = 0; d < DIM; ++d) {
    float f = row[d];
    __bf16 h = (__bf16)f;
    __bf16 l = (__bf16)(f - (float)h);
    hi[(size_t)k * DIM + d] = h;
    lo[(size_t)k * DIM + d] = l;
    s += f * f;
  }
  halfcsq[k] = 0.5f * s;
}

// Build one A fragment pair (hi/lo) for a 16-row tile, both K=32 sub-tiles.
// A 16x32 in contiguous K order (as in B): lanes 0-15 hold K=0..15, lanes 16-31 hold
// K=16..31; frag element e -> VGPR e/2 half e%2 (K ascending).
__device__ __forceinline__ void build_a_frags(const float* __restrict__ xrow,
                                              int h, v16bf aHi[2], v16bf aLo[2]) {
#pragma unroll
  for (int kt = 0; kt < 2; ++kt) {
    const float* p0 = xrow + kt * 32 + h * 16;       // K .. K+7
    const float* p1 = xrow + kt * 32 + h * 16 + 8;   // K+8 .. K+15
    v4f x0 = *(const v4f*)(p0);
    v4f x1 = *(const v4f*)(p0 + 4);
    v4f x2 = *(const v4f*)(p1);
    v4f x3 = *(const v4f*)(p1 + 4);
    float f[16] = {x0[0], x0[1], x0[2], x0[3], x1[0], x1[1], x1[2], x1[3],
                   x2[0], x2[1], x2[2], x2[3], x3[0], x3[1], x3[2], x3[3]};
    v16bf hfrag, lfrag;
#pragma unroll
    for (int j = 0; j < 16; ++j) {
      float fv = f[j];
      __bf16 hb = (__bf16)fv;
      hfrag[j] = hb;
      lfrag[j] = (__bf16)(fv - (float)hb);
    }
    aHi[kt] = hfrag;
    aLo[kt] = lfrag;
  }
}

// ---------------------------------------------------------------------------
// Main: 8 waves/block; each wave owns 32 rows (2 row-tiles, 2 acc chains).
// ---------------------------------------------------------------------------
__launch_bounds__(256)
__global__ void vq_argmin_kernel(const float* __restrict__ X,
                                 const float* __restrict__ CB,
                                 const __bf16* __restrict__ BH,
                                 const __bf16* __restrict__ BL,
                                 const float* __restrict__ halfcsq,
                                 float* __restrict__ outIdx,
                                 float* __restrict__ outResid,
                                 float* __restrict__ outEmb) {
  __shared__ float sHcsq[K_CODES];               // 4 KB bias table in LDS

  const int tid  = threadIdx.x;
  const int lane = tid & 31;
  const int wave = tid >> 5;
  const int rowBase = blockIdx.x * 256 + wave * 32;
  const int m = lane & 15;   // A: row within tile / B,C: column within tile
  const int h = lane >> 4;   // lane-half selects K sub-range per ISA layout

  // ---- Stage 0.5*||c||^2 into LDS with CDNA5 async-to-LDS (ASYNCcnt). ----
  // Low 32 bits of a flat LDS pointer == LDS byte address (aperture map).
  {
    uint32_t lofs = (uint32_t)(uintptr_t)(&sHcsq[tid * 4]);
    uint64_t ga   = (uint64_t)(uintptr_t)(halfcsq + tid * 4);
    asm volatile("global_load_async_to_lds_b128 %0, %1, off"
                 :: "v"(lofs), "v"(ga) : "memory");
  }

  // ---- Build A fragments for both row-tiles (overlaps async stage). ------
  v16bf aHi0[2], aLo0[2], aHi1[2], aLo1[2];
  build_a_frags(X + (size_t)(rowBase + m) * DIM,      h, aHi0, aLo0);
  build_a_frags(X + (size_t)(rowBase + 16 + m) * DIM, h, aHi1, aLo1);

  asm volatile("s_wait_asynccnt 0x0" ::: "memory");
  __syncthreads();

  // Running argmax: 8 rows per lane per row-tile; column = tile*16 + m.
  float bestV[2][8];
  int   bestI[2][8];
#pragma unroll
  for (int rt = 0; rt < 2; ++rt)
#pragma unroll
    for (int i = 0; i < 8; ++i) { bestV[rt][i] = -3.4e38f; bestI[rt][i] = 0; }

  // -------- Sweep all 64 codebook column tiles ----------------------------
  for (int t = 0; t < 64; ++t) {
    const int code = t * 16 + m;                 // this lane's column (code id)
    const size_t cbase = (size_t)code << 6;      // code*DIM

    if (t + 1 < 64) {                            // global_prefetch_b8 next tile
      __builtin_prefetch(BH + ((size_t)((t + 1) * 16 + m) << 6), 0, 3);
      __builtin_prefetch(BL + ((size_t)((t + 1) * 16 + m) << 6), 0, 3);
    }

    // Seed accumulators with -0.5*||c||^2 (bias hidden under WMMA latency).
    const float nhc = -sHcsq[code];              // ds_load_b32, conflict-free
    v8f acc0 = {nhc, nhc, nhc, nhc, nhc, nhc, nhc, nhc};
    v8f acc1 = acc0;

#pragma unroll
    for (int kt = 0; kt < 2; ++kt) {
      // B 32x16 bf16, contiguous K order as in A: lanes 0-15 hold K=0..15, lanes 16-31 K=16..31 of
      // this K-tile, column = lane&15 -> 32 contiguous bytes of a code row.
      v16bf bh = *(const v16bf*)(BH + cbase + kt * 32 + h * 16);
      v16bf bl = *(const v16bf*)(BL + cbase + kt * 32 + h * 16);
      const v16bf aH0 = (kt == 0) ? aHi0[0] : aHi0[1];
      const v16bf aL0 = (kt == 0) ? aLo0[0] : aLo0[1];
      const v16bf aH1 = (kt == 0) ? aHi1[0] : aHi1[1];
      const v16bf aL1 = (kt == 0) ? aLo1[0] : aLo1[1];
      // Two independent chains, interleaved for co-execution.
      acc0 = __builtin_amdgcn_wmma_f32_16x16x32_bf16(false, aH0, false, bh,
                                                     (short)0, acc0, false, false);
      acc1 = __builtin_amdgcn_wmma_f32_16x16x32_bf16(false, aH1, false, bh,
                                                     (short)0, acc1, false, false);
      acc0 = __builtin_amdgcn_wmma_f32_16x16x32_bf16(false, aH0, false, bl,
                                                     (short)0, acc0, false, false);
      acc1 = __builtin_amdgcn_wmma_f32_16x16x32_bf16(false, aH1, false, bl,
                                                     (short)0, acc1, false, false);
      acc0 = __builtin_amdgcn_wmma_f32_16x16x32_bf16(false, aL0, false, bh,
                                                     (short)0, acc0, false, false);
      acc1 = __builtin_amdgcn_wmma_f32_16x16x32_bf16(false, aL1, false, bh,
                                                     (short)0, acc1, false, false);
    }

    // Strict '>' keeps the earliest (smallest) code on ties within a lane.
#pragma unroll
    for (int i = 0; i < 8; ++i) {               // C layout: VGPR i = row i+8*h
      float v0 = acc0[i];
      bool b0 = v0 > bestV[0][i];
      bestV[0][i] = b0 ? v0 : bestV[0][i];
      bestI[0][i] = b0 ? code : bestI[0][i];
      float v1 = acc1[i];
      bool b1 = v1 > bestV[1][i];
      bestV[1][i] = b1 ? v1 : bestV[1][i];
      bestI[1][i] = b1 ? code : bestI[1][i];
    }
  }

  // -------- Cross-lane argmax over the 16 column lanes per row-half -------
#pragma unroll
  for (int rt = 0; rt < 2; ++rt)
#pragma unroll
    for (int i = 0; i < 8; ++i) {
      float v = bestV[rt][i];
      int   bi = bestI[rt][i];
#pragma unroll
      for (int off = 1; off < 16; off <<= 1) {  // stays within each 16-lane half
        float ov = __shfl_xor(v, off, 32);
        int   oi = __shfl_xor(bi, off, 32);
        if (ov > v || (ov == v && oi < bi)) { v = ov; bi = oi; }
      }
      bestV[rt][i] = v;
      bestI[rt][i] = bi;
    }

  // -------- Broadcast winners; write index, gather code, residual ---------
#pragma unroll
  for (int r = 0; r < 32; ++r) {
    const int rt  = r >> 4;                     // row-tile
    const int src = (r & 8) ? 16 : 0;           // rows 0-7 in lanes 0-15, 8-15 in 16-31
    const int idx = __shfl(bestI[rt][r & 7], src, 32);
    const int row = rowBase + r;
    if (lane == r) outIdx[row] = (float)idx;    // int index stored as float in d_out

    const float* xr = X + (size_t)row * DIM;
    const float* cr = CB + (size_t)idx * DIM;
    const int d = lane * 2;                     // 32 lanes x 2 floats = 64
    v2f xv = *(const v2f*)(xr + d);
    v2f cv = *(const v2f*)(cr + d);
    v2f rv = {xv[0] - cv[0], xv[1] - cv[1]};
    *(v2f*)(outEmb + (size_t)row * DIM + d) = cv;
    *(v2f*)(outResid + (size_t)row * DIM + d) = rv;
  }
}

// ---------------------------------------------------------------------------
extern "C" void kernel_launch(void* const* d_in, const int* in_sizes, int n_in,
                              void* d_out, int out_size, void* d_ws, size_t ws_size,
                              hipStream_t stream) {
  const float* X  = (const float*)d_in[0];   // previous_residual [N, D] f32
  const float* CB = (const float*)d_in[1];   // codebook_embeddings [K, D] f32

  // Workspace layout: hi (128KB) | lo (128KB) | halfcsq (4KB)
  __bf16* hi      = (__bf16*)d_ws;
  __bf16* lo      = (__bf16*)((char*)d_ws + (size_t)K_CODES * DIM * 2);
  float*  halfcsq = (float*) ((char*)d_ws + (size_t)K_CODES * DIM * 4);

  prep_codebook<<<(K_CODES + 255) / 256, 256, 0, stream>>>(CB, hi, lo, halfcsq);

  // d_out: [N] index (as float) | [N*D] next_residual | [N*D] codebook_embedding
  float* outIdx   = (float*)d_out;
  float* outResid = outIdx + N_PTS;
  float* outEmb   = outResid + (size_t)N_PTS * DIM;

  vq_argmin_kernel<<<N_PTS / 256, 256, 0, stream>>>(X, CB, hi, lo, halfcsq,
                                                    outIdx, outResid, outEmb);
}
